// Interval_Spline_Flow_20572893348084
// MI455X (gfx1250) — matrix operand order repaired and checked by reading, not compiled, not run on hardware
//
#include <hip/hip_runtime.h>
#include <hip/hip_bf16.h>

typedef _Float16 f16;
typedef __attribute__((ext_vector_type(16))) _Float16 v16h;
typedef __attribute__((ext_vector_type(8)))  float    v8f;

#define B_TOT   65536
#define H_DIM   512
#define KPAD1   96          // 16 (x, masked) + 64 (xc) + 16 zero pad
#define NT_HEADS 97         // 32 width bins + 32 height bins + 33 deriv bins
#define ROWS_WG 64
#define WAVES   4
#define DERIV_BIAS 0.54132485461292f   // log(e-1)

// workspace layout (f16 element offsets)
#define OFF_WCAT 0
#define SZ_WCAT  (H_DIM * KPAD1)                 // 49152
#define OFF_W2   (OFF_WCAT + SZ_WCAT)
#define SZ_W2    (H_DIM * H_DIM)                 // 262144
#define OFF_WW   (OFF_W2 + SZ_W2)
#define OFF_WH   (OFF_WW + SZ_W2)
#define OFF_WD   (OFF_WH + SZ_W2)
#define SZ_WD    (33 * 16 * H_DIM)               // 270336
#define WS_ELEMS (OFF_WD + SZ_WD)                // 1105920 f16 = ~2.2MB

// LDS layout (bytes), two time-overlapped regions:
//   phases 0-2: [A1: 4*16*96 f16 = 12288][H1: 4*16*512 f16 = 65536]   (77824 B)
//   phase  3  : [LGwh: 16*64*16 f16 = 32768][LGd: 16*33*16 f32 = 33792] (66560 B)
//   persistent: [H2: 64*512 f16 = 65536]
#define REGION0_B 77824
#define SMEM_B    (REGION0_B + ROWS_WG * H_DIM * 2)   // 143360 -> 2 WGs/WGP

// ---------------- prep: apply MADE masks, pack fp16 weights ----------------
__global__ void prep_weights(const float* __restrict__ W1, const float* __restrict__ W1c,
                             const float* __restrict__ W2, const float* __restrict__ Ww,
                             const float* __restrict__ Wh, const float* __restrict__ Wd,
                             f16* __restrict__ ws) {
    int i = blockIdx.x * 256 + threadIdx.x;
    if (i >= WS_ELEMS) return;
    float v;
    if (i < OFF_W2) {                                  // Wcat: [512][96] = [W1*m1 | W1c | 0]
        int o = i / KPAD1, c = i % KPAD1;
        int degh = (o % 15) + 1;
        if (c < 16)      v = (degh >= c + 1) ? W1[o * 16 + c] : 0.f;
        else if (c < 80) v = W1c[o * 64 + (c - 16)];
        else             v = 0.f;
    } else if (i < OFF_WW) {                           // W2 * m2
        int j = i - OFF_W2; int o = j >> 9, k = j & 511;
        v = (((o % 15) + 1) >= ((k % 15) + 1)) ? W2[j] : 0.f;
    } else if (i < OFF_WH) {                           // Ww * mo
        int j = i - OFF_WW; int o = j >> 9, k = j & 511;
        v = (((o & 15) + 1) > ((k % 15) + 1)) ? Ww[j] : 0.f;
    } else if (i < OFF_WD) {                           // Wh * mo
        int j = i - OFF_WH; int o = j >> 9, k = j & 511;
        v = (((o & 15) + 1) > ((k % 15) + 1)) ? Wh[j] : 0.f;
    } else {                                           // Wd * mod (528 rows)
        int j = i - OFF_WD; int o = j >> 9, k = j & 511;
        v = (((o & 15) + 1) > ((k % 15) + 1)) ? Wd[j] : 0.f;
    }
    ws[i] = (f16)v;
}

// ---------------- fragment loaders (CDNA5 16-bit WMMA layouts) ----------------
// A 16x32: lanes 0-15 (M=lane): elems 0-7 = K k0..k0+7, elems 8-15 = K k0+16..k0+23
//          lanes 16-31: +8 on both K ranges.
__device__ __forceinline__ v16h load_a(const f16* panel, int ldk, int k0, int lane) {
    int m = lane & 15, hi = lane >> 4;
    const f16* p = panel + m * ldk + k0 + hi * 8;
    union { uint4 u[2]; v16h v; } r;
    r.u[0] = *(const uint4*)(p);
    r.u[1] = *(const uint4*)(p + 16);
    return r.v;
}
// B 32x16: lane n holds column N=n&15, same K layout as A: elems 0-7 = K k0+h*8..+7, elems 8-15 = K k0+16+h*8..+7 (h = n>=16).
// Our B column N is row (n0+N) of the row-major masked weight matrix.
__device__ __forceinline__ v16h load_b(const f16* W, int ldk, int n0, int k0, int lane) {
    int n = n0 + (lane & 15), hi = lane >> 4;
    const f16* p = W + n * ldk + k0 + hi * 8;
    union { uint4 u[2]; v16h v; } r;
    r.u[0] = *(const uint4*)(p);
    r.u[1] = *(const uint4*)(p + 16);
    return r.v;
}

#define WMMA_F16(A, B, C) \
    __builtin_amdgcn_wmma_f32_16x16x32_f16(false, (A), false, (B), (short)0, (C), false, false)

// ---------------- fused MLP + spline ----------------
__global__ __launch_bounds__(128, 1)
void spline_flow_fused(const float* __restrict__ x,  const float* __restrict__ xc,
                       const float* __restrict__ b1, const float* __restrict__ b2,
                       const float* __restrict__ bw, const float* __restrict__ bh,
                       const float* __restrict__ bd,
                       const f16* __restrict__ ws,   float* __restrict__ outp) {
    extern __shared__ char smem[];
    // region 0 (time-multiplexed)
    f16*   A1   = (f16*)smem;                      // phases 0-2
    f16*   H1   = A1 + WAVES * 16 * KPAD1;         // phases 1-2
    f16*   LGwh = (f16*)smem;                      // phase 3: [16 r][64 t][16 d] f16
    float* LGd  = (float*)(smem + 32768);          // phase 3: [16 r][33 j][16 d] f32
    // persistent
    f16*   H2   = (f16*)(smem + REGION0_B);        // [64][512] f16

    const f16* Wcat = ws + OFF_WCAT;
    const f16* W2m  = ws + OFF_W2;
    const f16* Wwm  = ws + OFF_WW;
    const f16* Whm  = ws + OFF_WH;
    const f16* Wdm  = ws + OFF_WD;

    const int tid = threadIdx.x, wave = tid >> 5, lane = tid & 31;
    const int laneN = lane & 15, mb = (lane >> 4) * 8;
    const int row0  = blockIdx.x * ROWS_WG;
    const int wrow0 = row0 + wave * 16;

    // Phase 0: stage [x | xc | 0] fp16 A panel (per-wave 16x96, row major)
    f16* a1 = A1 + wave * 16 * KPAD1;
    for (int i = lane; i < 16 * KPAD1; i += 32) {
        int r = i / KPAD1, c = i % KPAD1;
        float v = (c < 16) ? x[(wrow0 + r) * 16 + c]
                : (c < 80) ? xc[(wrow0 + r) * 64 + (c - 16)] : 0.f;
        a1[i] = (f16)v;
    }
    __syncthreads();

    // Phase 1: h1 = relu([x|xc] @ Wcat^T + b1)   (K = 96, 3 WMMA per tile)
    f16* h1 = H1 + wave * 16 * H_DIM;
    {
        v16h af0 = load_a(a1, KPAD1, 0,  lane);
        v16h af1 = load_a(a1, KPAD1, 32, lane);
        v16h af2 = load_a(a1, KPAD1, 64, lane);
        for (int nt = 0; nt < 32; nt++) {
            float bias = b1[nt * 16 + laneN];
            v8f acc;
            #pragma unroll
            for (int i = 0; i < 8; i++) acc[i] = bias;
            acc = WMMA_F16(af0, load_b(Wcat, KPAD1, nt * 16, 0,  lane), acc);
            acc = WMMA_F16(af1, load_b(Wcat, KPAD1, nt * 16, 32, lane), acc);
            acc = WMMA_F16(af2, load_b(Wcat, KPAD1, nt * 16, 64, lane), acc);
            #pragma unroll
            for (int i = 0; i < 8; i++)
                h1[(mb + i) * H_DIM + nt * 16 + laneN] = (f16)fmaxf(acc[i], 0.f);
        }
    }
    __syncthreads();

    // Phase 2: h2 = relu(h1 @ W2m^T + b2)   (K = 512, A cached in 128 VGPRs)
    {
        v16h af[16];
        #pragma unroll
        for (int kb = 0; kb < 16; kb++) af[kb] = load_a(h1, H_DIM, kb * 32, lane);
        f16* h2 = H2 + wave * 16 * H_DIM;
        for (int nt = 0; nt < 32; nt++) {
            float bias = b2[nt * 16 + laneN];
            v8f acc;
            #pragma unroll
            for (int i = 0; i < 8; i++) acc[i] = bias;
            #pragma unroll
            for (int kb = 0; kb < 16; kb++)
                acc = WMMA_F16(af[kb], load_b(W2m, H_DIM, nt * 16, kb * 32, lane), acc);
            #pragma unroll
            for (int i = 0; i < 8; i++)
                h2[(mb + i) * H_DIM + nt * 16 + laneN] = (f16)fmaxf(acc[i], 0.f);
        }
    }
    __syncthreads();   // H2 visible to all; A1/H1 dead -> region 0 becomes logits

    // Phase 3: per 16-row block, all waves cooperatively compute the 97 head
    // tiles (tile t = logits of bin t for all 16 dims), then fp32 spline.
    for (int m = 0; m < 4; m++) {
        v16h af[16];
        const f16* h2m = H2 + m * 16 * H_DIM;
        #pragma unroll
        for (int kb = 0; kb < 16; kb++) af[kb] = load_a(h2m, H_DIM, kb * 32, lane);

        for (int t = wave; t < NT_HEADS; t += WAVES) {
            const f16* W; const float* bv; int tl;
            if (t < 32)      { W = Wwm; bv = bw; tl = t; }
            else if (t < 64) { W = Whm; bv = bh; tl = t - 32; }
            else             { W = Wdm; bv = bd; tl = t - 64; }
            float bias = bv[tl * 16 + laneN];
            v8f acc;
            #pragma unroll
            for (int i = 0; i < 8; i++) acc[i] = bias;
            #pragma unroll
            for (int kb = 0; kb < 16; kb++)
                acc = WMMA_F16(af[kb], load_b(W, H_DIM, tl * 16, kb * 32, lane), acc);
            if (t < 64) {          // width/height softmax logits -> f16
                #pragma unroll
                for (int i = 0; i < 8; i++)
                    LGwh[((mb + i) * 64 + t) * 16 + laneN] = (f16)acc[i];
            } else {               // derivative (softplus) logits -> f32
                #pragma unroll
                for (int i = 0; i < 8; i++)
                    LGd[((mb + i) * 33 + (t - 64)) * 16 + laneN] = acc[i];
            }
        }
        __syncthreads();

        // spline: 256 (row, dim) pairs over 128 threads
        for (int p = tid; p < 256; p += 128) {
            int r = p >> 4, d = p & 15;
            const f16*   Lw = LGwh + (r * 64) * 16 + d;   // uw at k*16, uh at (32+k)*16
            const float* Ld = LGd  + (r * 33) * 16 + d;   // ud at j*16
            float mw = -1e30f, mh = -1e30f;
            for (int k = 0; k < 32; k++) {
                mw = fmaxf(mw, (float)Lw[k * 16]);
                mh = fmaxf(mh, (float)Lw[(32 + k) * 16]);
            }
            float sw = 0.f, sh = 0.f;
            for (int k = 0; k < 32; k++) {
                sw += __expf((float)Lw[k * 16] - mw);
                sh += __expf((float)Lw[(32 + k) * 16] - mh);
            }
            const float cf = 1.f - 1e-3f * 32.f;
            float aw = cf / sw, ah = cf / sh;
            int gb = row0 + m * 16 + r;
            float xv = x[gb * 16 + d];

            // widths: incremental cumsum + searchsorted
            float c = 0.f; int idx = 0; float in_cw = -1.f, in_w = 2.f;
            for (int k = 0; k < 32; k++) {
                float w  = 1e-3f + aw * __expf((float)Lw[k * 16] - mw);
                float cn = c + w;
                float cl = (k == 0)  ? -1.f : (2.f * c  - 1.f);
                float cr = (k == 31) ?  1.f : (2.f * cn - 1.f);
                if (xv >= cl) { idx = k; in_cw = cl; in_w = cr - cl; }
                c = cn;
            }
            // heights at the found bin
            c = 0.f; float in_ch = -1.f, in_h = 2.f;
            for (int k = 0; k < 32; k++) {
                float h  = 1e-3f + ah * __expf((float)Lw[(32 + k) * 16] - mh);
                float cn = c + h;
                if (k == idx) {
                    float cl = (k == 0)  ? -1.f : (2.f * c  - 1.f);
                    float cr = (k == 31) ?  1.f : (2.f * cn - 1.f);
                    in_ch = cl; in_h = cr - cl;
                }
                c = cn;
            }
            // derivatives (softplus, stable) at idx, idx+1
            float z0 = Ld[idx * 16] + DERIV_BIAS;
            float z1 = Ld[(idx + 1) * 16] + DERIV_BIAS;
            float d0 = 1e-3f + ((z0 > 20.f) ? z0 : log1pf(__expf(z0)));
            float d1 = 1e-3f + ((z1 > 20.f) ? z1 : log1pf(__expf(z1)));

            float delta = in_h / in_w;
            float theta = (xv - in_cw) / in_w;
            float t1m   = theta * (1.f - theta);
            float denom = delta + (d0 + d1 - 2.f * delta) * t1m;
            float numer = in_h * (delta * theta * theta + d0 * t1m);
            float outv  = in_ch + numer / denom;
            float omt   = 1.f - theta;
            float dnum  = delta * delta * (d1 * theta * theta + 2.f * delta * t1m + d0 * omt * omt);
            float ldj   = __logf(dnum) - 2.f * __logf(denom);

            outp[gb * 16 + d] = outv;
            outp[B_TOT * 16 + gb * 16 + d] = ldj;
        }
        __syncthreads();
    }
}

extern "C" void kernel_launch(void* const* d_in, const int* in_sizes, int n_in,
                              void* d_out, int out_size, void* d_ws, size_t ws_size,
                              hipStream_t stream) {
    const float* x   = (const float*)d_in[0];
    const float* xc  = (const float*)d_in[1];
    const float* W1  = (const float*)d_in[2];
    const float* W1c = (const float*)d_in[3];
    const float* b1  = (const float*)d_in[4];
    const float* W2  = (const float*)d_in[5];
    const float* b2  = (const float*)d_in[6];
    const float* Ww  = (const float*)d_in[7];
    const float* bw  = (const float*)d_in[8];
    const float* Wh  = (const float*)d_in[9];
    const float* bh  = (const float*)d_in[10];
    const float* Wd  = (const float*)d_in[11];
    const float* bd  = (const float*)d_in[12];
    float* outp = (float*)d_out;
    f16*   ws   = (f16*)d_ws;

    prep_weights<<<(WS_ELEMS + 255) / 256, 256, 0, stream>>>(W1, W1c, W2, Ww, Wh, Wd, ws);

    hipFuncSetAttribute(reinterpret_cast<const void*>(&spline_flow_fused),
                        hipFuncAttributeMaxDynamicSharedMemorySize, (int)SMEM_B);
    spline_flow_fused<<<B_TOT / ROWS_WG, 128, SMEM_B, stream>>>(
        x, xc, b1, b2, bw, bh, bd, ws, outp);
}
